// MultiHeadAttention_26912265077338
// MI455X (gfx1250) — compile-verified, matrix operand order checked by reading, not run on hardware
//
#include <hip/hip_runtime.h>
#include <hip/hip_bf16.h>

typedef __attribute__((ext_vector_type(16))) __bf16 v16bf;
typedef __attribute__((ext_vector_type(8)))  __bf16 v8bf;
typedef __attribute__((ext_vector_type(8)))  float  v8f;

#define BATCH  4
#define SEQ    2048
#define DMODEL 1024
#define NHEADS 16
#define HDIM   64
#define MTOT   (BATCH * SEQ)   // 8192

// ---------------- common helpers ----------------

__device__ __forceinline__ v8f wmma_bf16(v16bf a, v16bf b, v8f c) {
  // D = A(16x32 bf16) * B(32x16 bf16) + C(16x16 f32)
  return __builtin_amdgcn_wmma_f32_16x16x32_bf16(
      /*neg_a=*/false, a, /*neg_b=*/false, b,
      /*c_mod=*/(short)0, c, /*reuse_a=*/false, /*reuse_b=*/false);
}

__device__ __forceinline__ v16bf pack16(v8bf lo, v8bf hi) {
  v16bf r;
#pragma unroll
  for (int i = 0; i < 8; ++i) { r[i] = lo[i]; r[i + 8] = hi[i]; }
  return r;
}

// ---------------- tiled GEMM (projections + output projection) ----------------
// C[M=8192, N=1024] = A[8192,1024] @ B[1024,1024] (+ bias)
// Block tile 128x128, K-step 32. 8 waves: 2 (M) x 4 (N), each wave 64x32 = 8 WMMA accs.
// Double-buffered LDS + register-staged global fetch: 1 barrier per K-step,
// global loads for tile k+1 overlap the 8 WMMAs of tile k.

#define BM 128
#define BN 128
#define BK 32
#define LDA_S 40   // LDS row stride (bf16 elems), padded to dodge bank conflicts
#define LDB_S 40

// MODE 0: A = f32 x, out = bf16 in [b][h][n][d] (QKV projection)
// MODE 1: A = bf16 ctx, out = f32 [M][N] + bias (output projection)
template <int MODE>
__global__ __launch_bounds__(256)
void mha_gemm_kernel(const float*  __restrict__ A32,
                     const __bf16* __restrict__ A16,
                     const float*  __restrict__ Bmat,
                     const float*  __restrict__ bias,
                     float*        __restrict__ outF,
                     __bf16*       __restrict__ outQ)
{
  __shared__ __bf16 As[2][BM * LDA_S];       // [m][k]
  __shared__ __bf16 Bs[2][BN * LDB_S];       // transposed: [n][k]

  const int tid   = threadIdx.x;
  const int lane  = tid & 31;
  const int wv    = tid >> 5;
  const int wm    = wv >> 2;                 // 0..1
  const int wn    = wv & 3;                  // 0..3
  const int mBase = blockIdx.x * BM;
  const int nBase = blockIdx.y * BN;
  const int lm    = lane & 15;
  const int k0    = (lane < 16) ? 0 : 8;     // ISA 16-bit fragment half-wave K offset
  const int mOff  = (lane < 16) ? 0 : 8;     // C-layout row offset

  v8f acc[4][2];
#pragma unroll
  for (int mt = 0; mt < 4; ++mt)
#pragma unroll
    for (int nt = 0; nt < 2; ++nt) acc[mt][nt] = {};

  const int arow = tid >> 1;                 // 0..127
  const int acol = (tid & 1) * 16;           // 0 / 16
  const int brow = tid >> 3;                 // 0..31 (k)
  const int bcol = (tid & 7) * 16;           // 0..112

  const float*  aSrc32 = (MODE == 0) ? &A32[(size_t)(mBase + arow) * DMODEL + acol] : nullptr;
  const __bf16* aSrc16 = (MODE == 1) ? &A16[(size_t)(mBase + arow) * DMODEL + acol] : nullptr;
  const float*  bSrc   = &Bmat[(size_t)brow * DMODEL + nBase + bcol];

  // register staging for the in-flight tile
  float4 fA0, fA1, fA2, fA3;
  v8bf   a16lo, a16hi;
  float4 fB[4];

  auto fetchTile = [&](int kb) {
    if (MODE == 0) {
      const float* s = aSrc32 + kb;
      fA0 = *(const float4*)(s + 0);  fA1 = *(const float4*)(s + 4);
      fA2 = *(const float4*)(s + 8);  fA3 = *(const float4*)(s + 12);
    } else {
      const __bf16* s = aSrc16 + kb;
      a16lo = *(const v8bf*)(s);      a16hi = *(const v8bf*)(s + 8);
    }
    const float* bs = bSrc + (size_t)kb * DMODEL;
#pragma unroll
    for (int j4 = 0; j4 < 4; ++j4) fB[j4] = *(const float4*)(bs + j4 * 4);
  };

  auto stageTile = [&](int buf) {
    v8bf lo, hi;
    if (MODE == 0) {
      lo[0]=(__bf16)fA0.x; lo[1]=(__bf16)fA0.y; lo[2]=(__bf16)fA0.z; lo[3]=(__bf16)fA0.w;
      lo[4]=(__bf16)fA1.x; lo[5]=(__bf16)fA1.y; lo[6]=(__bf16)fA1.z; lo[7]=(__bf16)fA1.w;
      hi[0]=(__bf16)fA2.x; hi[1]=(__bf16)fA2.y; hi[2]=(__bf16)fA2.z; hi[3]=(__bf16)fA2.w;
      hi[4]=(__bf16)fA3.x; hi[5]=(__bf16)fA3.y; hi[6]=(__bf16)fA3.z; hi[7]=(__bf16)fA3.w;
    } else {
      lo = a16lo; hi = a16hi;
    }
    v8bf* adst = (v8bf*)&As[buf][arow * LDA_S + acol];
    adst[0] = lo; adst[1] = hi;
#pragma unroll
    for (int j4 = 0; j4 < 4; ++j4) {
      const float4 f = fB[j4];
      __bf16* bdst = &Bs[buf][(bcol + j4 * 4) * LDB_S + brow];
      bdst[0 * LDB_S] = (__bf16)f.x;
      bdst[1 * LDB_S] = (__bf16)f.y;
      bdst[2 * LDB_S] = (__bf16)f.z;
      bdst[3 * LDB_S] = (__bf16)f.w;
    }
  };

  auto computeTile = [&](int buf) {
    v16bf Af[4], Bf[2];
#pragma unroll
    for (int mt = 0; mt < 4; ++mt) {
      const __bf16* p = &As[buf][(wm * 64 + mt * 16 + lm) * LDA_S + k0];
      Af[mt] = pack16(*(const v8bf*)p, *(const v8bf*)(p + 16));
    }
#pragma unroll
    for (int nt = 0; nt < 2; ++nt) {
      const __bf16* p = &Bs[buf][(wn * 32 + nt * 16 + lm) * LDB_S + k0];
      Bf[nt] = pack16(*(const v8bf*)p, *(const v8bf*)(p + 16));
    }
#pragma unroll
    for (int mt = 0; mt < 4; ++mt)
#pragma unroll
      for (int nt = 0; nt < 2; ++nt)
        acc[mt][nt] = wmma_bf16(Af[mt], Bf[nt], acc[mt][nt]);
  };

  constexpr int NK = DMODEL / BK;            // 32 K-steps
  fetchTile(0);
  stageTile(0);
#pragma unroll 1
  for (int kt = 0; kt < NK; ++kt) {
    __syncthreads();                         // buf[kt&1] now fully staged
    if (kt + 1 < NK) {
      fetchTile((kt + 1) * BK);              // global loads overlap the WMMAs below
      if (kt + 2 < NK) {                     // gfx1250 global_prefetch two tiles out
        const char* pa = (MODE == 0) ? (const char*)(aSrc32 + (kt + 2) * BK)
                                     : (const char*)(aSrc16 + (kt + 2) * BK);
        __builtin_prefetch(pa, 0, 3);
        __builtin_prefetch((const char*)(bSrc + (size_t)(kt + 2) * BK * DMODEL), 0, 3);
      }
    }
    computeTile(kt & 1);
    if (kt + 1 < NK) stageTile((kt + 1) & 1);
  }

  // ---- epilogue: C layout = VGPR i -> rows (i, i+8) split by half-wave, cols = lane%16 ----
#pragma unroll
  for (int mt = 0; mt < 4; ++mt) {
#pragma unroll
    for (int nt = 0; nt < 2; ++nt) {
      const int n = nBase + wn * 32 + nt * 16 + lm;
#pragma unroll
      for (int i = 0; i < 8; ++i) {
        const int m = mBase + wm * 64 + mt * 16 + mOff + i;
        const float val = acc[mt][nt][i];
        if (MODE == 0) {
          const int b = m >> 11, nn = m & (SEQ - 1);
          const int h = n >> 6,  dd = n & (HDIM - 1);
          outQ[(((size_t)(b * NHEADS + h) * SEQ) + nn) * HDIM + dd] = (__bf16)val;
        } else {
          outF[(size_t)m * DMODEL + n] = val + bias[n];
        }
      }
    }
  }
}

// ---------------- causal flash attention ----------------
// grid: (SEQ/128 q-blocks, BATCH*NHEADS). 8 waves, wave w owns q rows [qBase+16w, +16).
// Key blocks of 32 double-buffered in LDS; online softmax; P reshaped via LDS for PV WMMAs.

#define KT_S 72   // Kt row stride ([key][d])
#define VT_S 40   // Vt row stride ([d][key], transposed)
#define P_S  40   // P  row stride ([q][key]) per wave

__global__ __launch_bounds__(256)
void mha_attn_kernel(const __bf16* __restrict__ Q,
                     const __bf16* __restrict__ K,
                     const __bf16* __restrict__ V,
                     __bf16*       __restrict__ ctxOut)  // [b][n][D] bf16
{
  __shared__ __bf16 Kt[2][32 * KT_S];
  __shared__ __bf16 Vt[2][64 * VT_S];
  __shared__ __bf16 Pl[8 * 16 * P_S];

  const int tid   = threadIdx.x;
  const int lane  = tid & 31;
  const int wv    = tid >> 5;
  const int qBase = blockIdx.x * 128;
  const int bh    = blockIdx.y;
  const int lm    = lane & 15;
  const int k0    = (lane < 16) ? 0 : 8;
  const int mOff  = (lane < 16) ? 0 : 8;

  // persistent Q fragments (16 rows x 64 d -> two 16x32 A-fragments)
  const int qRow = qBase + wv * 16 + lm;
  const __bf16* qp = &Q[((size_t)bh * SEQ + qRow) * HDIM];
  v16bf Qf[2];
#pragma unroll
  for (int dh = 0; dh < 2; ++dh) {
    const __bf16* p = qp + dh * 32 + k0;
    Qf[dh] = pack16(*(const v8bf*)p, *(const v8bf*)(p + 16));
  }

  v8f ctx[4];
#pragma unroll
  for (int t = 0; t < 4; ++t) ctx[t] = {};
  float rowmax[8], rowsum[8];
#pragma unroll
  for (int i = 0; i < 8; ++i) { rowmax[i] = -__builtin_inff(); rowsum[i] = 0.0f; }

  const int ldKey = tid >> 3;                // 0..31
  const int ldD   = (tid & 7) * 8;           // 0..56
  const __bf16* kSrc = &K[((size_t)bh * SEQ + ldKey) * HDIM + ldD];
  const __bf16* vSrc = &V[((size_t)bh * SEQ + ldKey) * HDIM + ldD];

  v8bf kReg, vReg;
  auto fetchKV = [&](int kb) {
    kReg = *(const v8bf*)(kSrc + (size_t)kb * HDIM);
    vReg = *(const v8bf*)(vSrc + (size_t)kb * HDIM);
  };
  auto stageKV = [&](int buf) {
    *(v8bf*)&Kt[buf][ldKey * KT_S + ldD] = kReg;
#pragma unroll
    for (int j = 0; j < 8; ++j)
      Vt[buf][(ldD + j) * VT_S + ldKey] = vReg[j];
  };

  const int nBlk = blockIdx.x * 4 + 4;       // uniform causal key-block count
  fetchKV(0);
  stageKV(0);
#pragma unroll 1
  for (int blk = 0; blk < nBlk; ++blk) {
    const int kb  = blk * 32;
    const int cur = blk & 1;
    __syncthreads();                         // buf[cur] staged
    if (blk + 1 < nBlk) {
      fetchKV((blk + 1) * 32);               // overlap next K/V fetch with math
      if (blk + 2 < nBlk) {
        __builtin_prefetch((const char*)(kSrc + (size_t)(blk + 2) * 32 * HDIM), 0, 3);
        __builtin_prefetch((const char*)(vSrc + (size_t)(blk + 2) * 32 * HDIM), 0, 3);
      }
    }

    // ---- S = Q K^T : two 16-key sub-tiles, contraction over d=64 (2 WMMAs each) ----
    v8f S[2];
#pragma unroll
    for (int ks = 0; ks < 2; ++ks) {
      S[ks] = {};
#pragma unroll
      for (int dh = 0; dh < 2; ++dh) {
        const __bf16* p = &Kt[cur][(ks * 16 + lm) * KT_S + dh * 32 + k0];
        const v16bf Kf = pack16(*(const v8bf*)p, *(const v8bf*)(p + 16));
        S[ks] = wmma_bf16(Qf[dh], Kf, S[ks]);
      }
    }

    // ---- scale + causal mask + online softmax (rows live across a half-wave) ----
    float pva[8], pvb[8], bmax[8];
#pragma unroll
    for (int i = 0; i < 8; ++i) {
      const int qg = qBase + wv * 16 + mOff + i;
      float s0 = S[0][i] * 0.125f;           // 1/sqrt(64)
      float s1 = S[1][i] * 0.125f;
      s0 = ((kb + lm)      > qg) ? -__builtin_inff() : s0;
      s1 = ((kb + 16 + lm) > qg) ? -__builtin_inff() : s1;
      pva[i] = s0; pvb[i] = s1;
      bmax[i] = fmaxf(s0, s1);
    }
#pragma unroll
    for (int i = 0; i < 8; ++i)
#pragma unroll
      for (int off = 1; off < 16; off <<= 1)
        bmax[i] = fmaxf(bmax[i], __shfl_xor(bmax[i], off, 16));

    float fac[8];
#pragma unroll
    for (int i = 0; i < 8; ++i) {
      const float nm = fmaxf(rowmax[i], bmax[i]);
      fac[i] = __expf(rowmax[i] - nm);
      rowmax[i] = nm;
      const float p0 = __expf(pva[i] - nm);
      const float p1 = __expf(pvb[i] - nm);
      __bf16* pd = &Pl[(wv * 16 + mOff + i) * P_S];
      pd[lm]      = (__bf16)p0;              // reshape C-layout -> A-layout via LDS
      pd[16 + lm] = (__bf16)p1;
      float ps = p0 + p1;
#pragma unroll
      for (int off = 1; off < 16; off <<= 1)
        ps += __shfl_xor(ps, off, 16);
      rowsum[i] = rowsum[i] * fac[i] + ps;
    }
#pragma unroll
    for (int t = 0; t < 4; ++t)
#pragma unroll
      for (int i = 0; i < 8; ++i)
        ctx[t][i] *= fac[i];

    // ---- ctx += P[16x32] @ V[32x64] : 4 WMMAs across d-tiles ----
    v16bf Pf;
    {
      const __bf16* p = &Pl[(wv * 16 + lm) * P_S + k0];
      Pf = pack16(*(const v8bf*)p, *(const v8bf*)(p + 16));
    }
#pragma unroll
    for (int t = 0; t < 4; ++t) {
      const __bf16* p = &Vt[cur][(t * 16 + lm) * VT_S + k0];
      const v16bf Vf = pack16(*(const v8bf*)p, *(const v8bf*)(p + 16));
      ctx[t] = wmma_bf16(Pf, Vf, ctx[t]);
    }

    if (blk + 1 < nBlk) stageKV((blk + 1) & 1);
  }

  // ---- normalize and emit ctx in [b][n][h*64+d] so the output GEMM reads row-major ----
  const int b = bh >> 4;
  const int h = bh & 15;
#pragma unroll
  for (int t = 0; t < 4; ++t) {
#pragma unroll
    for (int i = 0; i < 8; ++i) {
      const int qg = qBase + wv * 16 + mOff + i;
      const float val = ctx[t][i] / rowsum[i];
      ctxOut[((size_t)b * SEQ + qg) * DMODEL + h * HDIM + t * 16 + lm] = (__bf16)val;
    }
  }
}

// ---------------- host-side launcher ----------------

extern "C" void kernel_launch(void* const* d_in, const int* in_sizes, int n_in,
                              void* d_out, int out_size, void* d_ws, size_t ws_size,
                              hipStream_t stream) {
  const float* x   = (const float*)d_in[0];
  const float* W_q = (const float*)d_in[1];
  const float* W_k = (const float*)d_in[2];
  const float* W_v = (const float*)d_in[3];
  const float* W_o = (const float*)d_in[4];
  const float* b_o = (const float*)d_in[5];
  float* out = (float*)d_out;

  // workspace: Q | K | V in [b][h][n][d] bf16 (16 MB each), ctx in [b][n][D] bf16 (16 MB)
  const size_t SLAB = (size_t)BATCH * NHEADS * SEQ * HDIM * sizeof(__bf16); // 16 MiB
  char* ws = (char*)d_ws;
  __bf16* q_ws   = (__bf16*)(ws);
  __bf16* k_ws   = (__bf16*)(ws + SLAB);
  __bf16* v_ws   = (__bf16*)(ws + 2 * SLAB);
  __bf16* ctx_ws = (__bf16*)(ws + 3 * SLAB);

  const dim3 gemmGrid(MTOT / BM, DMODEL / BN);    // 64 x 8
  const dim3 attnGrid(SEQ / 128, BATCH * NHEADS); // 16 x 64

  // QKV projections (f32 x -> bf16 heads layout)
  mha_gemm_kernel<0><<<gemmGrid, 256, 0, stream>>>(x, nullptr, W_q, nullptr, nullptr, q_ws);
  mha_gemm_kernel<0><<<gemmGrid, 256, 0, stream>>>(x, nullptr, W_k, nullptr, nullptr, k_ws);
  mha_gemm_kernel<0><<<gemmGrid, 256, 0, stream>>>(x, nullptr, W_v, nullptr, nullptr, v_ws);

  // causal flash attention
  mha_attn_kernel<<<attnGrid, 256, 0, stream>>>(q_ws, k_ws, v_ws, ctx_ws);

  // output projection (bf16 ctx @ f32 W_o + b_o -> f32 out)
  mha_gemm_kernel<1><<<gemmGrid, 256, 0, stream>>>(nullptr, ctx_ws, W_o, b_o, out, nullptr);
}
